// Attention_45887430590573
// MI455X (gfx1250) — compile-verified, matrix operand order checked by reading, not run on hardware
//
#include <hip/hip_runtime.h>
#include <hip/hip_bf16.h>

// ---------------------------------------------------------------------------
// Multi-head attention (B=8, N=2048, C=768, H=12, HD=64) for gfx1250.
// f16 operands + fp32 accumulation via V_WMMA_F32_16X16X32_F16.
// Flash-style fused softmax; 64-column chunks; lane-partial softmax denom
// (no ds_bpermute for the sum); SCALE folded into Q fragments.
// ---------------------------------------------------------------------------

typedef _Float16 v16h __attribute__((ext_vector_type(16)));
typedef _Float16 v8h  __attribute__((ext_vector_type(8)));
typedef float    v8f  __attribute__((ext_vector_type(8)));

#define BDIM  8
#define NTOK  2048
#define CDIM  768
#define HCNT  12
#define HDIM  64
#define QKVLD 2304          // 3*C
#define SCLH  ((_Float16)0.125f)   // 64^-0.5, exact in f16

// ---- WMMA fragment helpers -------------------------------------------------
// 16-bit A operand (16x32): lane L holds row (L&15); element i (0..15) is
// K = (i/8)*16 + (L>>4)*8 + (i%8)  ->  two contiguous 8-half (16B) runs.
// B operand (32x16) uses the same pattern over a column-major (transposed) src.
__device__ __forceinline__ v16h ld_frag(const _Float16* rowp, int k0, int lane) {
    int koff = k0 + ((lane >> 4) << 3);
    union { v16h v; v8h h[2]; } u;
    u.h[0] = *(const v8h*)(rowp + koff);
    u.h[1] = *(const v8h*)(rowp + koff + 16);
    return u.v;
}

__device__ __forceinline__ v8f wmma_f16(v16h a, v16h b, v8f c) {
    // (neg_a, A, neg_b, B, c_mod, C, reuse_a, reuse_b)
    return __builtin_amdgcn_wmma_f32_16x16x32_f16(false, a, false, b,
                                                  (short)0, c, false, false);
}

// reductions across the 16-lane half-group (rows of C-layout live there)
__device__ __forceinline__ float redmax16(float x) {
#pragma unroll
    for (int m = 1; m < 16; m <<= 1) x = fmaxf(x, __shfl_xor(x, m, 32));
    return x;
}
__device__ __forceinline__ float redsum16(float x) {
#pragma unroll
    for (int m = 1; m < 16; m <<= 1) x += __shfl_xor(x, m, 32);
    return x;
}

// ---- conversion / transpose kernels ---------------------------------------
__global__ void k_cvt_f16(const float* __restrict__ in, _Float16* __restrict__ out, int n) {
    int i = blockIdx.x * blockDim.x + threadIdx.x;
    if (i < n) out[i] = (_Float16)in[i];
}

// out[c][r] = in[r][c]   (in: rows x cols, f32) -> f16 transposed
__global__ void k_transpose_f16(const float* __restrict__ in, _Float16* __restrict__ out,
                                int rows, int cols) {
    int i = blockIdx.x * blockDim.x + threadIdx.x;
    if (i >= rows * cols) return;
    int r = i / cols, c = i % cols;
    out[(size_t)c * rows + r] = (_Float16)in[i];
}

// vT[b][h][d][n] = qkv[b][n][2][h][d]   (V slice, head-dim major)
__global__ void k_make_vT(const _Float16* __restrict__ qkv, _Float16* __restrict__ vT) {
    int i = blockIdx.x * blockDim.x + threadIdx.x;
    if (i >= BDIM * HCNT * HDIM * NTOK) return;
    int n  = i & (NTOK - 1);
    int d  = (i >> 11) & (HDIM - 1);
    int bh = i >> 17;
    int h  = bh % HCNT, b = bh / HCNT;
    vT[i] = qkv[((size_t)(b * NTOK + n)) * QKVLD + 2 * CDIM + h * HDIM + d];
}

// ---- generic GEMM: C[M,N] = A[M,K] @ Bt[N,K]^T  (A,Bt f16 row-major) -------
// 4 waves/block, each wave computes a 16x64 strip (4 N-tiles, reuse A frag).
template <bool F16OUT>
__global__ __launch_bounds__(128) void k_gemm_wmma(
    const _Float16* __restrict__ A, const _Float16* __restrict__ Bt,
    void* __restrict__ Cout, const float* __restrict__ bias,
    int K, int lda, int ldb, int ldc) {
    int lane  = threadIdx.x & 31;
    int wave  = threadIdx.x >> 5;
    int mBase = (blockIdx.y * 4 + wave) * 16;
    int nBase = blockIdx.x * 64;

    v8f c[4];
#pragma unroll
    for (int t = 0; t < 4; ++t)
#pragma unroll
        for (int r = 0; r < 8; ++r) c[t][r] = 0.f;

    const _Float16* arow = A + (size_t)(mBase + (lane & 15)) * lda;
    for (int k0 = 0; k0 < K; k0 += 32) {
        v16h a = ld_frag(arow, k0, lane);
#pragma unroll
        for (int t = 0; t < 4; ++t) {
            const _Float16* brow = Bt + (size_t)(nBase + t * 16 + (lane & 15)) * ldb;
            v16h b = ld_frag(brow, k0, lane);
            c[t] = wmma_f16(a, b, c[t]);
        }
    }

    int rb = (lane >> 4) << 3;   // C-layout: lane holds rows rb..rb+7
    int col = lane & 15;
#pragma unroll
    for (int t = 0; t < 4; ++t)
#pragma unroll
        for (int r = 0; r < 8; ++r) {
            size_t idx = (size_t)(mBase + rb + r) * ldc + nBase + t * 16 + col;
            if (F16OUT) ((_Float16*)Cout)[idx] = (_Float16)c[t][r];
            else        ((float*)Cout)[idx]    = c[t][r] + bias[nBase + t * 16 + col];
        }
}

// ---- fused flash attention -------------------------------------------------
// One wave per block; block = (bh, 16-query-row tile). Online softmax over
// 32 iterations of 64 key columns. P bounced through a 16x64 f16 LDS tile.
__global__ __launch_bounds__(32) void k_attn(
    const _Float16* __restrict__ qkv, const _Float16* __restrict__ vT,
    _Float16* __restrict__ out) {
    __shared__ _Float16 lds[16 * 64];
    int lane = threadIdx.x;
    int bh = blockIdx.x;
    int b = bh / HCNT, h = bh % HCNT;
    int rowBase = blockIdx.y * 16;

    // Q fragments (K = HD = 64 -> two 32-chunks), with softmax scale folded in
    const _Float16* qrow = qkv + ((size_t)(b * NTOK) + rowBase + (lane & 15)) * QKVLD + h * HDIM;
    v16h aq0 = ld_frag(qrow, 0, lane);
    v16h aq1 = ld_frag(qrow, 32, lane);
#pragma unroll
    for (int i = 0; i < 16; ++i) { aq0[i] *= SCLH; aq1[i] *= SCLH; }

    const _Float16* kbase = qkv + (size_t)(b * NTOK) * QKVLD + CDIM + h * HDIM; // + n*QKVLD
    const _Float16* vtb   = vT + (size_t)bh * HDIM * NTOK;                      // + d*NTOK

    float mrow[8], lpart[8];   // lpart = per-LANE partial denominator
    v8f acc[4];
#pragma unroll
    for (int r = 0; r < 8; ++r) { mrow[r] = -1.0e30f; lpart[r] = 0.f; }
#pragma unroll
    for (int t = 0; t < 4; ++t)
#pragma unroll
        for (int r = 0; r < 8; ++r) acc[t][r] = 0.f;

    int rb  = (lane >> 4) << 3;
    int col = lane & 15;

    for (int j = 0; j < NTOK; j += 64) {
        // --- scores: four 16x16 tiles covering columns j .. j+63 ---
        v8f s[4];
#pragma unroll
        for (int t = 0; t < 4; ++t)
#pragma unroll
            for (int r = 0; r < 8; ++r) s[t][r] = 0.f;
#pragma unroll
        for (int t = 0; t < 4; ++t) {
            const _Float16* kp = kbase + (size_t)(j + t * 16 + (lane & 15)) * QKVLD;
            v16h bk0 = ld_frag(kp, 0, lane);
            s[t] = wmma_f16(aq0, bk0, s[t]);
            v16h bk1 = ld_frag(kp, 32, lane);
            s[t] = wmma_f16(aq1, bk1, s[t]);
        }
        // prefetch next chunk's K rows into the cache hierarchy
        if (j + 64 < NTOK) {
            __builtin_prefetch(kbase + (size_t)(j + 64 + (lane & 15)) * QKVLD, 0, 3);
            __builtin_prefetch(kbase + (size_t)(j + 112 + (lane & 15)) * QKVLD, 0, 3);
        }

        // --- online softmax per row (max-reduce only; sum stays per-lane) ---
        float fac[8];
#pragma unroll
        for (int r = 0; r < 8; ++r) {
            float xm = fmaxf(fmaxf(s[0][r], s[1][r]), fmaxf(s[2][r], s[3][r]));
            float mn = fmaxf(mrow[r], redmax16(xm));
            float f  = __expf(mrow[r] - mn);
            float psum = 0.f;
#pragma unroll
            for (int t = 0; t < 4; ++t) {
                float p = __expf(s[t][r] - mn);
                psum += p;
                lds[(rb + r) * 64 + t * 16 + col] = (_Float16)p;
            }
            lpart[r] = lpart[r] * f + psum;
            mrow[r]  = mn;
            fac[r]   = f;
        }
#pragma unroll
        for (int t = 0; t < 4; ++t)
#pragma unroll
            for (int r = 0; r < 8; ++r) acc[t][r] *= fac[r];

        __syncthreads();   // single-wave block: lowers to waitcnt + S_NOP

        // --- PV: P(16x64) @ V(64x64); vT rows = head-dims (column-major V) ---
        v16h ap0 = ld_frag(&lds[(lane & 15) * 64], 0, lane);
        v16h ap1 = ld_frag(&lds[(lane & 15) * 64], 32, lane);
#pragma unroll
        for (int t = 0; t < 4; ++t) {
            const _Float16* vrow = vtb + (size_t)(t * 16 + (lane & 15)) * NTOK + j;
            v16h bv0 = ld_frag(vrow, 0, lane);
            acc[t] = wmma_f16(ap0, bv0, acc[t]);
            v16h bv1 = ld_frag(vrow, 32, lane);
            acc[t] = wmma_f16(ap1, bv1, acc[t]);
        }
        if (j + 64 < NTOK) {
            __builtin_prefetch(vtb + (size_t)((lane & 15)) * NTOK + j + 64, 0, 3);
            __builtin_prefetch(vtb + (size_t)(48 + (lane & 15)) * NTOK + j + 64, 0, 3);
        }
        __syncthreads();
    }

    // --- finalize: single cross-lane sum for the denominator, then store ---
#pragma unroll
    for (int r = 0; r < 8; ++r) {
        float inv = 1.f / redsum16(lpart[r]);
#pragma unroll
        for (int t = 0; t < 4; ++t) {
            size_t idx = ((size_t)(b * NTOK) + rowBase + rb + r) * CDIM
                         + h * HDIM + t * 16 + col;
            out[idx] = (_Float16)(acc[t][r] * inv);
        }
    }
}

// ---------------------------------------------------------------------------
extern "C" void kernel_launch(void* const* d_in, const int* in_sizes, int n_in,
                              void* d_out, int out_size, void* d_ws, size_t ws_size,
                              hipStream_t stream) {
    const float* x      = (const float*)d_in[0];  // [B*N, C]
    const float* w_qkv  = (const float*)d_in[1];  // [C, 3C]
    const float* w_proj = (const float*)d_in[2];  // [C, C]
    const float* b_proj = (const float*)d_in[3];  // [C]
    float* out = (float*)d_out;                   // [B*N, C]

    const size_t MR = (size_t)BDIM * NTOK;        // 16384 rows
    _Float16* xh     = (_Float16*)d_ws;                       // [16384, 768]
    _Float16* wqkvT  = xh     + MR * CDIM;                    // [2304, 768]
    _Float16* wprojT = wqkvT  + (size_t)QKVLD * CDIM;         // [768, 768]
    _Float16* qkv_h  = wprojT + (size_t)CDIM * CDIM;          // [16384, 2304]
    _Float16* vTh    = qkv_h  + MR * QKVLD;                   // [96, 64, 2048]
    _Float16* aoh    = vTh    + MR * CDIM;                    // [16384, 768]

    // 1) convert x to f16; transpose+convert weights
    {
        int n = (int)(MR * CDIM);
        k_cvt_f16<<<(n + 255) / 256, 256, 0, stream>>>(x, xh, n);
    }
    {
        int n = CDIM * QKVLD;
        k_transpose_f16<<<(n + 255) / 256, 256, 0, stream>>>(w_qkv, wqkvT, CDIM, QKVLD);
    }
    {
        int n = CDIM * CDIM;
        k_transpose_f16<<<(n + 255) / 256, 256, 0, stream>>>(w_proj, wprojT, CDIM, CDIM);
    }

    // 2) QKV projection: qkv_h[16384,2304] = xh @ wqkvT^T  (f16 out)
    k_gemm_wmma<true><<<dim3(QKVLD / 64, (int)(MR / 64)), 128, 0, stream>>>(
        xh, wqkvT, qkv_h, nullptr, CDIM, CDIM, CDIM, QKVLD);

    // 3) V transpose per head: vT[bh][d][n]
    {
        int n = (int)(MR * CDIM);
        k_make_vT<<<(n + 255) / 256, 256, 0, stream>>>(qkv_h, vTh);
    }

    // 4) fused flash attention -> aoh[16384, 768] (heads merged, f16)
    k_attn<<<dim3(BDIM * HCNT, NTOK / 16), 32, 0, stream>>>(qkv_h, vTh, aoh);

    // 5) output projection + bias: out = aoh @ wprojT^T + b_proj (f32 out)
    k_gemm_wmma<false><<<dim3(CDIM / 64, (int)(MR / 64)), 128, 0, stream>>>(
        aoh, wprojT, out, b_proj, CDIM, CDIM, CDIM, CDIM);
}
